// MultiHeadSelfAttention_13511967113396
// MI455X (gfx1250) — matrix operand order repaired and checked by reading, not compiled, not run on hardware
//
#include <hip/hip_runtime.h>
#include <math.h>

#define NUM_LABELS 8
#define D_MODEL 1024
#define NUM_HEADS 16
#define D_HEAD 64
#define BATCH 4
#define SEQ 2048
#define QKV3 (3 * D_MODEL)

typedef __attribute__((ext_vector_type(16))) _Float16 v16h;
typedef __attribute__((ext_vector_type(8)))  float    v8f;
typedef _Float16 h16;

// ---------------------------------------------------------------------------
// WMMA fragment loaders (layouts per CDNA5 ISA §7.12.2, wave32).
// A (16x32 f16): lane l holds row M=l&15; halves e=0..7 -> K=kb+e,
//                e=8..15 -> K=16+kb+(e-8), kb = (l<16 ? 0 : 8).
// B (32x16 f16): lane l holds col N=l&15; same K order as A (kb=(l<16?0:8)).
// C/D (16x16 f32): VGPR r -> row r + (l<16?0:8), col = l&15.
// ---------------------------------------------------------------------------
__device__ __forceinline__ v16h frag_a(const h16* p, int ld) {
  const int lane = threadIdx.x & 31;
  const h16* r = p + (lane & 15) * ld + ((lane & 16) ? 8 : 0);
  union { v16h v; float4 f[2]; } u;
  u.f[0] = *(const float4*)(r);
  u.f[1] = *(const float4*)(r + 16);
  return u.v;
}

__device__ __forceinline__ v16h frag_b(const h16* p, int ld) {
  const int lane = threadIdx.x & 31;
  const h16* r = p + (lane & 15) * ld + ((lane & 16) ? 8 : 0);
  union { v16h v; float4 f[2]; } u;
  u.f[0] = *(const float4*)(r);
  u.f[1] = *(const float4*)(r + 16);
  return u.v;
}

__device__ __forceinline__ v8f wmma16(v16h a, v16h b, v8f c) {
  return __builtin_amdgcn_wmma_f32_16x16x32_f16(false, a, false, b, (short)0, c,
                                                false, false);
}

// ---------------------------------------------------------------------------
// 1) fp32 -> f16 convert for X
// ---------------------------------------------------------------------------
__global__ void k_cvt_x(const float* __restrict__ X, h16* __restrict__ Xh, int n) {
  int i = blockIdx.x * blockDim.x + threadIdx.x;
  if (i < n) Xh[i] = (h16)X[i];
}

// 2) Wqkv [8][1024][3072] fp32 -> Wqkvt [8][3072][1024] f16 (K-contiguous B frags)
__global__ void k_tr_wqkv(const float* __restrict__ W, h16* __restrict__ Wt) {
  int d = blockIdx.x * blockDim.x + threadIdx.x;  // 0..1023
  int o = blockIdx.y;                             // 0..3071
  int l = blockIdx.z;                             // 0..7
  Wt[((size_t)l * QKV3 + o) * D_MODEL + d] =
      (h16)W[((size_t)l * D_MODEL + d) * QKV3 + o];
}

// 3) Wproj [1024][1024] fp32 -> Wprojt [o][d] f16
__global__ void k_tr_wproj(const float* __restrict__ W, h16* __restrict__ Wt) {
  int d = blockIdx.x * blockDim.x + threadIdx.x;
  int o = blockIdx.y;
  Wt[(size_t)o * D_MODEL + d] = (h16)W[(size_t)d * D_MODEL + o];
}

// ---------------------------------------------------------------------------
// 4) Label-routed QKV projection, ping-pong software pipeline (unroll-by-2 so
// the two fragment buffers alternate roles -> no v_mov hand-off copies).
// One wave32 per block; __launch_bounds__(32,1) lifts the VGPR cap.
// Grid: (col_tile 0..47 [64 cols], row_tile 0..511 [16 tokens], label 0..7)
// ---------------------------------------------------------------------------
__global__ void __launch_bounds__(32, 1)
k_qkv(const h16* __restrict__ Xh, const int* __restrict__ L,
      const h16* __restrict__ Wt, const float* __restrict__ bqkv,
      h16* __restrict__ Qh, h16* __restrict__ Kh, h16* __restrict__ Vt) {
  const int l = blockIdx.z;
  const int rt = blockIdx.y;
  const int b = rt >> 7;               // 128 row tiles per batch
  const int row0 = (rt & 127) << 4;
  const int col0 = blockIdx.x << 6;

  const int* Lb = L + b * SEQ + row0;
  unsigned lm = 0;
#pragma unroll
  for (int i = 0; i < 16; ++i) lm |= 1u << Lb[i];
  if (!((lm >> l) & 1u)) return;       // wave-uniform exit (EXEC stays full)

  const h16* A = Xh + ((size_t)b * SEQ + row0) * D_MODEL;
  const h16* Bw = Wt + ((size_t)l * QKV3 + col0) * D_MODEL;

  v8f acc[4] = {};
  v16h a0, a1, b0[4], b1[4];
  a0 = frag_a(A, D_MODEL);
#pragma unroll
  for (int c = 0; c < 4; ++c)
    b0[c] = frag_b(Bw + (size_t)(c * 16) * D_MODEL, D_MODEL);

  int k0 = 32;
  for (; k0 + 32 < D_MODEL; k0 += 64) {   // k0 = 32..928, 15 iterations
    __builtin_prefetch(A + k0 + 224 + (threadIdx.x & 15) * D_MODEL, 0, 3);
    a1 = frag_a(A + k0, D_MODEL);
#pragma unroll
    for (int c = 0; c < 4; ++c)
      b1[c] = frag_b(Bw + (size_t)(c * 16) * D_MODEL + k0, D_MODEL);
#pragma unroll
    for (int c = 0; c < 4; ++c) acc[c] = wmma16(a0, b0[c], acc[c]);
    a0 = frag_a(A + k0 + 32, D_MODEL);
#pragma unroll
    for (int c = 0; c < 4; ++c)
      b0[c] = frag_b(Bw + (size_t)(c * 16) * D_MODEL + k0 + 32, D_MODEL);
#pragma unroll
    for (int c = 0; c < 4; ++c) acc[c] = wmma16(a1, b1[c], acc[c]);
  }
  // epilogue: k0 == 992; buf0 holds k=960, stage final panel then drain both
  a1 = frag_a(A + k0, D_MODEL);
#pragma unroll
  for (int c = 0; c < 4; ++c)
    b1[c] = frag_b(Bw + (size_t)(c * 16) * D_MODEL + k0, D_MODEL);
#pragma unroll
  for (int c = 0; c < 4; ++c) acc[c] = wmma16(a0, b0[c], acc[c]);
#pragma unroll
  for (int c = 0; c < 4; ++c) acc[c] = wmma16(a1, b1[c], acc[c]);

  const int lane = threadIdx.x & 31;
  const int n = lane & 15;
  const int hi = (lane & 16) ? 8 : 0;

#pragma unroll
  for (int c = 0; c < 4; ++c) {
    float bs = bqkv[l * QKV3 + col0 + c * 16 + n];
#pragma unroll
    for (int r = 0; r < 8; ++r) acc[c][r] += bs;
  }

#pragma unroll
  for (int r = 0; r < 8; ++r) {
    int M = r + hi;
    if (Lb[M] != l) continue;          // row belongs to another label
    int tok = row0 + M;
#pragma unroll
    for (int c = 0; c < 4; ++c) {
      int col = col0 + c * 16 + n;
      float v = acc[c][r];
      if (col < D_MODEL) {
        Qh[((size_t)b * SEQ + tok) * D_MODEL + col] = (h16)v;
      } else if (col < 2 * D_MODEL) {
        Kh[((size_t)b * SEQ + tok) * D_MODEL + (col - D_MODEL)] = (h16)v;
      } else {
        Vt[((size_t)b * D_MODEL + (col - 2 * D_MODEL)) * SEQ + tok] = (h16)v;
      }
    }
  }
}

// ---------------------------------------------------------------------------
// 5) Flash attention, one wave per (32 q-rows, head): two 16-row Q tiles share
//    every K/V fragment. KV in blocks of 32: S = Q K^T (8 WMMA), online
//    softmax (f32, xor-shuffle row reductions), P staged through LDS
//    (C-layout -> A-layout), O += P V (8 WMMA).
// ---------------------------------------------------------------------------
__global__ void __launch_bounds__(32, 1)
k_attn(const h16* __restrict__ Qh, const h16* __restrict__ Kh,
       const h16* __restrict__ Vt, h16* __restrict__ Oh) {
  const int bh = blockIdx.y;
  const int b = bh >> 4, h = bh & 15;
  const int row0 = blockIdx.x << 5;        // 32 q rows per wave
  const int lane = threadIdx.x & 31;
  const int n = lane & 15;
  const int hi = (lane & 16) ? 8 : 0;

  __shared__ __align__(16) h16 Pbuf[2][16 * 32];

  const h16* Qb = Qh + ((size_t)b * SEQ + row0) * D_MODEL + h * D_HEAD;
  const h16* Kb = Kh + (size_t)b * SEQ * D_MODEL + h * D_HEAD;
  const h16* Vb = Vt + ((size_t)b * D_MODEL + h * D_HEAD) * SEQ;

  v16h qa[2][2];
#pragma unroll
  for (int t = 0; t < 2; ++t) {
    qa[t][0] = frag_a(Qb + (size_t)t * 16 * D_MODEL, D_MODEL);
    qa[t][1] = frag_a(Qb + (size_t)t * 16 * D_MODEL + 32, D_MODEL);
  }

  v8f acc[2][4] = {};
  float mrow[2][8], lrow[2][8];
#pragma unroll
  for (int t = 0; t < 2; ++t)
#pragma unroll
    for (int r = 0; r < 8; ++r) { mrow[t][r] = -INFINITY; lrow[t][r] = 0.f; }
  const float sc = 0.125f;  // 1/sqrt(64)

  for (int kv0 = 0; kv0 < SEQ; kv0 += 32) {
    const h16* K0 = Kb + (size_t)kv0 * D_MODEL;
    // speculative prefetch of the next 32 K rows (dropped silently at the end)
    __builtin_prefetch(K0 + (size_t)(32 + lane) * D_MODEL, 0, 3);
    v16h kb[4];
    kb[0] = frag_b(K0, D_MODEL);
    kb[1] = frag_b(K0 + 32, D_MODEL);
    kb[2] = frag_b(K0 + (size_t)16 * D_MODEL, D_MODEL);
    kb[3] = frag_b(K0 + (size_t)16 * D_MODEL + 32, D_MODEL);

#pragma unroll
    for (int t = 0; t < 2; ++t) {
      v8f s0 = {}, s1 = {};
      s0 = wmma16(qa[t][0], kb[0], s0);
      s0 = wmma16(qa[t][1], kb[1], s0);
      s1 = wmma16(qa[t][0], kb[2], s1);
      s1 = wmma16(qa[t][1], kb[3], s1);

#pragma unroll
      for (int r = 0; r < 8; ++r) {
        float v0 = s0[r] * sc, v1 = s1[r] * sc;
        float mx = fmaxf(v0, v1);
        mx = fmaxf(mx, __shfl_xor(mx, 1));
        mx = fmaxf(mx, __shfl_xor(mx, 2));
        mx = fmaxf(mx, __shfl_xor(mx, 4));
        mx = fmaxf(mx, __shfl_xor(mx, 8));   // row max within 16-lane half
        float mn = fmaxf(mrow[t][r], mx);
        float p0 = __expf(v0 - mn);
        float p1 = __expf(v1 - mn);
        float rs = p0 + p1;
        rs += __shfl_xor(rs, 1);
        rs += __shfl_xor(rs, 2);
        rs += __shfl_xor(rs, 4);
        rs += __shfl_xor(rs, 8);             // row sum within 16-lane half
        float alpha = __expf(mrow[t][r] - mn);  // first iter: exp(-inf)=0
        lrow[t][r] = lrow[t][r] * alpha + rs;
        mrow[t][r] = mn;
#pragma unroll
        for (int c = 0; c < 4; ++c) acc[t][c][r] *= alpha;
        int M = r + hi;
        Pbuf[t][M * 32 + n] = (h16)p0;
        Pbuf[t][M * 32 + 16 + n] = (h16)p1;
      }
    }
    __syncthreads();
    v16h pa0 = frag_a(Pbuf[0], 32);  // LDS reads, A layout
    v16h pa1 = frag_a(Pbuf[1], 32);
#pragma unroll
    for (int c = 0; c < 4; ++c) {
      v16h vb = frag_b(Vb + (size_t)(c * 16) * SEQ + kv0, SEQ);
      acc[0][c] = wmma16(pa0, vb, acc[0][c]);
      acc[1][c] = wmma16(pa1, vb, acc[1][c]);
    }
    __syncthreads();
  }

#pragma unroll
  for (int t = 0; t < 2; ++t)
#pragma unroll
    for (int r = 0; r < 8; ++r) {
      float inv = 1.f / lrow[t][r];
      int tok = row0 + t * 16 + r + hi;
#pragma unroll
      for (int c = 0; c < 4; ++c)
        Oh[((size_t)b * SEQ + tok) * D_MODEL + h * D_HEAD + c * 16 + n] =
            (h16)(acc[t][c][r] * inv);
    }
}

// ---------------------------------------------------------------------------
// 6) Output projection: out = Oh @ Wproj + bproj, fp32 stores to d_out.
//    Same ping-pong pipeline as k_qkv.
// ---------------------------------------------------------------------------
__global__ void __launch_bounds__(32, 1)
k_proj(const h16* __restrict__ Oh, const h16* __restrict__ Wt,
       const float* __restrict__ bproj, float* __restrict__ out) {
  const int row0 = blockIdx.y << 4;   // over B*N = 8192 tokens
  const int col0 = blockIdx.x << 6;
  const h16* A = Oh + (size_t)row0 * D_MODEL;
  const h16* Bw = Wt + (size_t)col0 * D_MODEL;

  v8f acc[4] = {};
  v16h a0, a1, b0[4], b1[4];
  a0 = frag_a(A, D_MODEL);
#pragma unroll
  for (int c = 0; c < 4; ++c)
    b0[c] = frag_b(Bw + (size_t)(c * 16) * D_MODEL, D_MODEL);

  int k0 = 32;
  for (; k0 + 32 < D_MODEL; k0 += 64) {   // k0 = 32..928, 15 iterations
    __builtin_prefetch(A + k0 + 224 + (threadIdx.x & 15) * D_MODEL, 0, 3);
    a1 = frag_a(A + k0, D_MODEL);
#pragma unroll
    for (int c = 0; c < 4; ++c)
      b1[c] = frag_b(Bw + (size_t)(c * 16) * D_MODEL + k0, D_MODEL);
#pragma unroll
    for (int c = 0; c < 4; ++c) acc[c] = wmma16(a0, b0[c], acc[c]);
    a0 = frag_a(A + k0 + 32, D_MODEL);
#pragma unroll
    for (int c = 0; c < 4; ++c)
      b0[c] = frag_b(Bw + (size_t)(c * 16) * D_MODEL + k0 + 32, D_MODEL);
#pragma unroll
    for (int c = 0; c < 4; ++c) acc[c] = wmma16(a1, b1[c], acc[c]);
  }
  a1 = frag_a(A + k0, D_MODEL);
#pragma unroll
  for (int c = 0; c < 4; ++c)
    b1[c] = frag_b(Bw + (size_t)(c * 16) * D_MODEL + k0, D_MODEL);
#pragma unroll
  for (int c = 0; c < 4; ++c) acc[c] = wmma16(a0, b0[c], acc[c]);
#pragma unroll
  for (int c = 0; c < 4; ++c) acc[c] = wmma16(a1, b1[c], acc[c]);

  const int lane = threadIdx.x & 31;
  const int n = lane & 15;
  const int hi = (lane & 16) ? 8 : 0;
#pragma unroll
  for (int c = 0; c < 4; ++c) {
    float bs = bproj[col0 + c * 16 + n];
#pragma unroll
    for (int r = 0; r < 8; ++r)
      out[((size_t)(row0 + r + hi)) * D_MODEL + col0 + c * 16 + n] =
          acc[c][r] + bs;
  }
}

// ---------------------------------------------------------------------------
extern "C" void kernel_launch(void* const* d_in, const int* in_sizes, int n_in,
                              void* d_out, int out_size, void* d_ws,
                              size_t ws_size, hipStream_t stream) {
  const float* X = (const float*)d_in[0];
  const int* L = (const int*)d_in[1];
  const float* Wqkv = (const float*)d_in[2];
  const float* bqkv = (const float*)d_in[3];
  const float* Wproj = (const float*)d_in[4];
  const float* bproj = (const float*)d_in[5];
  float* out = (float*)d_out;

  char* p = (char*)d_ws;
  h16* Xh = (h16*)p;      p += (size_t)BATCH * SEQ * D_MODEL * sizeof(h16);
  h16* Wqkvt = (h16*)p;   p += (size_t)NUM_LABELS * QKV3 * D_MODEL * sizeof(h16);
  h16* Wprojt = (h16*)p;  p += (size_t)D_MODEL * D_MODEL * sizeof(h16);
  h16* Qh = (h16*)p;      p += (size_t)BATCH * SEQ * D_MODEL * sizeof(h16);
  h16* Kh = (h16*)p;      p += (size_t)BATCH * SEQ * D_MODEL * sizeof(h16);
  h16* Vt = (h16*)p;      p += (size_t)BATCH * D_MODEL * SEQ * sizeof(h16);
  h16* Oh = (h16*)p;      p += (size_t)BATCH * SEQ * D_MODEL * sizeof(h16);

  const int nx = BATCH * SEQ * D_MODEL;
  k_cvt_x<<<(nx + 255) / 256, 256, 0, stream>>>(X, Xh, nx);
  k_tr_wqkv<<<dim3(D_MODEL / 256, QKV3, NUM_LABELS), 256, 0, stream>>>(Wqkv, Wqkvt);
  k_tr_wproj<<<dim3(D_MODEL / 256, D_MODEL), 256, 0, stream>>>(Wproj, Wprojt);
  k_qkv<<<dim3(QKV3 / 64, BATCH * SEQ / 16, NUM_LABELS), 32, 0, stream>>>(
      Xh, L, Wqkvt, bqkv, Qh, Kh, Vt);
  k_attn<<<dim3(SEQ / 32, BATCH * NUM_HEADS), 32, 0, stream>>>(Qh, Kh, Vt, Oh);
  k_proj<<<dim3(D_MODEL / 64, BATCH * SEQ / 16), 32, 0, stream>>>(
      Oh, Wprojt, bproj, out);
}
